// MobileViTBlockV2_16922171146677
// MI455X (gfx1250) — compile-verified, matrix operand order checked by reading, not run on hardware
//
#include <hip/hip_runtime.h>
#include <hip/hip_bf16.h>
#include <math.h>

// MobileViTBlockV2 for gfx1250 (MI455X, wave32, WMMA + Tensor Data Mover).
// 1x1 convs = bf16 WMMA GEMMs (v_wmma_f32_16x16x32_bf16). A (weight) tiles are
// DMA'd into LDS via tensor_load_to_lds (TENSORcnt); B (activation) tiles are
// stored transposed [n][k] so WMMA fragment reads are contiguous K-pairs.
// Workspace: ~226 MB (layout in kernel_launch).

typedef __bf16 bf16;
typedef __attribute__((ext_vector_type(16))) __bf16 v16bf;
typedef __attribute__((ext_vector_type(8)))  float  v8f;
typedef __attribute__((ext_vector_type(4)))  unsigned int u32x4;
typedef __attribute__((ext_vector_type(8)))  int i32x8;
typedef __attribute__((ext_vector_type(4)))  int i32x4;

#define SS    65536         // B * P * N = 16*4*1024 (GEMM column count)
#define EPSV  1e-5f

#if defined(__HIP_DEVICE_COMPILE__) && \
    __has_builtin(__builtin_amdgcn_tensor_load_to_lds) && \
    __has_builtin(__builtin_amdgcn_s_wait_tensorcnt)
#define USE_TDM 1
#else
#define USE_TDM 0
#endif

#if USE_TDM
// Issue a TDM 2D tile load: tile_d1 rows of tile_d0 contiguous bf16 elements,
// global row stride = stride0 elements, packed row-major into LDS at lds_off.
// D# layout per CDNA5 ISA ch.8 (group0: count/lds_addr/global_addr/type=2;
// group1: data_size=2B + tensor/tile dims + dim0 stride).
__device__ __forceinline__ void tdm_load_2d(unsigned lds_off, const void* gaddr,
                                            unsigned tile_d0, unsigned tile_d1,
                                            unsigned stride0) {
    unsigned long long ga = (unsigned long long)(uintptr_t)gaddr;
    u32x4 g0;
    g0[0] = 1u;                                   // count=1 (valid), user mode
    g0[1] = lds_off;                              // LDS byte address
    g0[2] = (unsigned)(ga & 0xFFFFFFFFu);         // global_addr[31:0]
    g0[3] = (unsigned)((ga >> 32) & 0x1FFFFFFu) | (2u << 30);  // addr[56:32] | type=2
    i32x8 g1;
    g1[0] = (int)(1u << 16);                      // data_size=1 (2 bytes)
    g1[1] = (int)((tile_d0 & 0xFFFFu) << 16);     // tensor_dim0[15:0] (== tile)
    g1[2] = (int)((tile_d0 >> 16) | ((tile_d1 & 0xFFFFu) << 16)); // dim0 hi | tensor_dim1 lo
    g1[3] = (int)((tile_d1 >> 16) | (tile_d0 << 16));            // dim1 hi | tile_dim0
    g1[4] = (int)(tile_d1 & 0xFFFFu);             // tile_dim1, tile_dim2=0
    g1[5] = (int)stride0;                         // tensor_dim0_stride[31:0]
    g1[6] = 0;
    g1[7] = 0;
    i32x4 z4; z4[0] = 0; z4[1] = 0; z4[2] = 0; z4[3] = 0;
    i32x8 z8;
    #pragma unroll
    for (int j = 0; j < 8; ++j) z8[j] = 0;
    __builtin_amdgcn_tensor_load_to_lds(g0, g1, z4, z4, z8, 0);
}
#endif

// ---------------------------------------------------------------------------
// Depthwise 3x3 (SAME) + BN + SiLU -> bf16 in patch layout:
// out[c, s],  s = ((b*4 + (h&1)*2+(w&1)) * 1024) + (h/2)*32 + (w/2)
// ---------------------------------------------------------------------------
__global__ void __launch_bounds__(256)
dwconv_bn_silu_kernel(const float* __restrict__ x, const float* __restrict__ w,
                      const float* __restrict__ g, const float* __restrict__ bb_,
                      const float* __restrict__ m, const float* __restrict__ vv,
                      bf16* __restrict__ out) {
    int idx = blockIdx.x * 256 + threadIdx.x;        // 16*256*64*64 total
    int wi = idx & 63;
    int h  = (idx >> 6) & 63;
    int c  = (idx >> 12) & 255;
    int b  = idx >> 20;
    const float* xp = x + (((size_t)(b * 256 + c)) << 12);
    const float* wp = w + c * 9;
    float acc = 0.f;
    #pragma unroll
    for (int i = 0; i < 3; ++i) {
        int hh = h + i - 1;
        if (hh < 0 || hh > 63) continue;
        #pragma unroll
        for (int j = 0; j < 3; ++j) {
            int ww = wi + j - 1;
            if (ww < 0 || ww > 63) continue;
            acc += xp[hh * 64 + ww] * wp[i * 3 + j];
        }
    }
    float scale = g[c] * rsqrtf(vv[c] + EPSV);
    float val = (acc - m[c]) * scale + bb_[c];
    val = val / (1.f + __expf(-val));                 // SiLU
    int s = ((b * 4 + (h & 1) * 2 + (wi & 1)) << 10) + ((h >> 1) << 5) + (wi >> 1);
    out[(size_t)c * SS + s] = (bf16)val;
}

__global__ void f2bf_kernel(const float* __restrict__ in, bf16* __restrict__ out, int n) {
    int i = blockIdx.x * 256 + threadIdx.x;
    if (i < n) out[i] = (bf16)in[i];
}

__global__ void zero_kernel(float* __restrict__ p, int n) {
    int i = blockIdx.x * blockDim.x + threadIdx.x;
    if (i < n) p[i] = 0.f;
}

// ---------------------------------------------------------------------------
// WMMA GEMM: out[M,S] = act(W[M,K] * X[K,S] + bias)
// Block tile 128(M) x 128(S), 8 waves as 4(M)x2(N); wave tile 32x64:
// 2 A-frags x 4 B-frags -> 8 v_wmma per K-step (K-step 32).
// A tile DMA'd into LDS via tensor_load_to_lds (wave 0); B tile stored
// transposed [n][k] so fragment reads are contiguous K-pairs (b32 pairs).
// OUTMODE: 0 = f32 store, 1 = f32 accumulate (residual), 2 = bf16 store
// ACT:     0 = none, 1 = SiLU, 2 = ReLU
// ---------------------------------------------------------------------------
template<int OUTMODE, int ACT>
__global__ void __launch_bounds__(256)
gemm_bf16_wmma(const bf16* __restrict__ W, const bf16* __restrict__ X,
               const float* __restrict__ bias, void* __restrict__ outp,
               int M, int K) {
    __shared__ bf16 As[128][32];   // [m][k] compact row-major (TDM layout)
    __shared__ bf16 Bs[128][34];   // [n][k] transposed, padded vs bank conflicts

    const int tid   = threadIdx.x;
    const int lane  = tid & 31;
    const int wave  = tid >> 5;
    const int waveM = wave >> 1;   // 0..3
    const int waveN = wave & 1;    // 0..1
    const int m0 = blockIdx.y * 128;
    const int s0 = blockIdx.x * 128;

#if USE_TDM
    const unsigned asOff = (unsigned)(uintptr_t)&As[0][0];
#endif

    v8f acc[2][4];
    #pragma unroll
    for (int mi = 0; mi < 2; ++mi)
        #pragma unroll
        for (int cg = 0; cg < 4; ++cg)
            #pragma unroll
            for (int r = 0; r < 8; ++r) acc[mi][cg][r] = 0.f;

    for (int k0 = 0; k0 < K; k0 += 32) {
        __syncthreads();   // previous iteration's LDS reads done before overwrite
#if USE_TDM
        if (wave == 0) {
            // A tile: 128 rows x 32 k, row stride K -> async DMA to LDS
            tdm_load_2d(asOff, W + (size_t)m0 * K + k0, 32u, 128u, (unsigned)K);
        }
#else
        {   // A tile: 128x32, 16 elems/thread
            int row = tid >> 1;
            int kl  = (tid & 1) * 16;
            const bf16* src = W + (size_t)(m0 + row) * K + k0 + kl;
            #pragma unroll
            for (int j = 0; j < 16; ++j) As[row][kl + j] = src[j];
        }
#endif
        {   // B tile: 32x128 global, stored transposed [n][k]; 16 elems/thread
            int kl = tid >> 3;             // 0..31
            int nb = (tid & 7) * 16;       // 0..112
            const bf16* src = X + (size_t)(k0 + kl) * SS + s0 + nb;
            #pragma unroll
            for (int j = 0; j < 16; ++j) Bs[nb + j][kl] = src[j];
        }
#if USE_TDM
        if (wave == 0) __builtin_amdgcn_s_wait_tensorcnt(0);
#endif
        __syncthreads();

        const int half = lane >> 4;
        // 2 A fragments (16x32 each) - contiguous K-pairs from [m][k] layout
        v16bf afrag[2];
        #pragma unroll
        for (int mi = 0; mi < 2; ++mi) {
            const int rowA = waveM * 32 + mi * 16 + (lane & 15);
            #pragma unroll
            for (int v = 0; v < 8; ++v) {
                int kb = (v >> 2) * 16 + half * 8 + (v & 3) * 2;
                afrag[mi][2 * v]     = As[rowA][kb];
                afrag[mi][2 * v + 1] = As[rowA][kb + 1];
            }
        }
        // 4 B fragments (32x16 each) - contiguous K-pairs from [n][k] layout,
        // reused across both A fragments
        #pragma unroll
        for (int cg = 0; cg < 4; ++cg) {
            const int col = waveN * 64 + cg * 16 + (lane & 15);
            v16bf bfrag;
            #pragma unroll
            for (int v = 0; v < 8; ++v) {
                int kb = (v >> 2) * 16 + half * 8 + (v & 3) * 2;
                bfrag[2 * v]     = Bs[col][kb];
                bfrag[2 * v + 1] = Bs[col][kb + 1];
            }
            #pragma unroll
            for (int mi = 0; mi < 2; ++mi) {
                acc[mi][cg] = __builtin_amdgcn_wmma_f32_16x16x32_bf16(
                    false, afrag[mi], false, bfrag, (short)0, acc[mi][cg], false, false);
            }
        }
    }

    // Epilogue. C/D layout: VGPR r -> row = half*8 + r, col = lane&15.
    const int half = lane >> 4;
    #pragma unroll
    for (int mi = 0; mi < 2; ++mi) {
        #pragma unroll
        for (int cg = 0; cg < 4; ++cg) {
            const int rowb = m0 + waveM * 32 + mi * 16 + half * 8;
            const int col  = s0 + waveN * 64 + cg * 16 + (lane & 15);
            #pragma unroll
            for (int r = 0; r < 8; ++r) {
                float v = acc[mi][cg][r];
                int row = rowb + r;
                if (bias) v += bias[row];
                if (ACT == 1) v = v / (1.f + __expf(-v));
                else if (ACT == 2) v = fmaxf(v, 0.f);
                size_t o = (size_t)row * SS + col;
                if (OUTMODE == 0)      ((float*)outp)[o] = v;
                else if (OUTMODE == 1) ((float*)outp)[o] += v;
                else                   ((bf16*)outp)[o] = (bf16)v;
            }
        }
    }
}

// ---------------------------------------------------------------------------
// GroupNorm(1 group) stats: per-batch sum/sumsq over 256*4096 elems.
// ---------------------------------------------------------------------------
__global__ void __launch_bounds__(256)
gn_stats_kernel(const float* __restrict__ P, float* __restrict__ stats) {
    int b = blockIdx.x;
    int t = threadIdx.x;
    float s = 0.f, s2 = 0.f;
    for (int f = blockIdx.y * 256 + t; f < 256 * 4096; f += 64 * 256) {
        int d  = f >> 12;
        int cl = f & 4095;
        float v = P[(size_t)d * SS + b * 4096 + cl];
        s += v; s2 += v * v;
    }
    __shared__ float rs[256], rs2[256];
    rs[t] = s; rs2[t] = s2;
    __syncthreads();
    for (int off = 128; off; off >>= 1) {
        if (t < off) { rs[t] += rs[t + off]; rs2[t] += rs2[t + off]; }
        __syncthreads();
    }
    if (t == 0) {
        atomicAdd(&stats[b * 2], rs[0]);
        atomicAdd(&stats[b * 2 + 1], rs2[0]);
    }
}

__global__ void __launch_bounds__(256)
gn_norm_kernel(const float* __restrict__ P, const float* __restrict__ stats,
               const float* __restrict__ g, const float* __restrict__ bia,
               bf16* __restrict__ out) {
    int idx = blockIdx.x * 256 + threadIdx.x;     // 256*65536
    int d = idx >> 16;
    int s = idx & 65535;
    int b = s >> 12;
    const float cnt = 256.f * 4096.f;
    float mu   = stats[b * 2] / cnt;
    float var  = stats[b * 2 + 1] / cnt - mu * mu;
    float rstd = rsqrtf(var + EPSV);
    float v = (P[idx] - mu) * rstd * g[d] + bia[d];
    out[idx] = (bf16)v;
}

// ---------------------------------------------------------------------------
// q = wq . x_norm (+bias), softmax over N=1024 within each (b,p).
// ---------------------------------------------------------------------------
__global__ void __launch_bounds__(256)
qsoftmax_kernel(const bf16* __restrict__ Xn, const bf16* __restrict__ wq,
                const float* __restrict__ qb, float* __restrict__ scores) {
    int bp = blockIdx.x;
    int t  = threadIdx.x;
    const int base = bp << 10;
    float q[4];
    #pragma unroll
    for (int j = 0; j < 4; ++j) q[j] = qb[0];
    for (int c = 0; c < 256; ++c) {
        float wc = (float)wq[c];
        const bf16* xr = Xn + (size_t)c * SS + base + t * 4;
        q[0] += wc * (float)xr[0];
        q[1] += wc * (float)xr[1];
        q[2] += wc * (float)xr[2];
        q[3] += wc * (float)xr[3];
    }
    __shared__ float red[256];
    float mx = fmaxf(fmaxf(q[0], q[1]), fmaxf(q[2], q[3]));
    red[t] = mx; __syncthreads();
    for (int off = 128; off; off >>= 1) {
        if (t < off) red[t] = fmaxf(red[t], red[t + off]);
        __syncthreads();
    }
    mx = red[0]; __syncthreads();
    float e[4], ssum = 0.f;
    #pragma unroll
    for (int j = 0; j < 4; ++j) { e[j] = __expf(q[j] - mx); ssum += e[j]; }
    red[t] = ssum; __syncthreads();
    for (int off = 128; off; off >>= 1) {
        if (t < off) red[t] += red[t + off];
        __syncthreads();
    }
    float inv = 1.f / red[0];
    #pragma unroll
    for (int j = 0; j < 4; ++j) scores[base + t * 4 + j] = e[j] * inv;
}

// ctx[bp, d] = sum_n k[d, bp*1024+n] * scores[bp, n].  grid(64,32), wave per d.
__global__ void __launch_bounds__(256)
ctx_kernel(const float* __restrict__ Kf, const float* __restrict__ scores,
           float* __restrict__ ctx) {
    int bp   = blockIdx.x;
    int wave = threadIdx.x >> 5;
    int lane = threadIdx.x & 31;
    int d = blockIdx.y * 8 + wave;
    const float* kr = Kf + (size_t)d * SS + (bp << 10);
    const float* sc = scores + (bp << 10);
    float s = 0.f;
    for (int n = lane; n < 1024; n += 32) s += kr[n] * sc[n];
    #pragma unroll
    for (int off = 16; off; off >>= 1) s += __shfl_down(s, off, 32);
    if (lane == 0) ctx[bp * 256 + d] = s;
}

// out_bf[d, s] = relu(v)[d, s] * ctx[bp(s), d]
__global__ void __launch_bounds__(256)
vmul_kernel(const float* __restrict__ Vr, const float* __restrict__ ctx,
            bf16* __restrict__ out) {
    int idx = blockIdx.x * 256 + threadIdx.x;
    int d = idx >> 16;
    int s = idx & 65535;
    int bp = s >> 10;
    out[idx] = (bf16)(Vr[idx] * ctx[bp * 256 + d]);
}

// Fold patches back to NCHW + BatchNorm
__global__ void __launch_bounds__(256)
bnfold_kernel(const float* __restrict__ G, const float* __restrict__ g,
              const float* __restrict__ b_, const float* __restrict__ m,
              const float* __restrict__ v, float* __restrict__ out) {
    int idx = blockIdx.x * 256 + threadIdx.x;       // 16*256*64*64
    int w = idx & 63;
    int h = (idx >> 6) & 63;
    int c = (idx >> 12) & 255;
    int b = idx >> 20;
    int s = ((b * 4 + (h & 1) * 2 + (w & 1)) << 10) + ((h >> 1) << 5) + (w >> 1);
    float sc = g[c] * rsqrtf(v[c] + EPSV);
    out[idx] = (G[(size_t)c * SS + s] - m[c]) * sc + b_[c];
}

// ---------------------------------------------------------------------------
extern "C" void kernel_launch(void* const* d_in, const int* in_sizes, int n_in,
                              void* d_out, int out_size, void* d_ws, size_t ws_size,
                              hipStream_t stream) {
    const float* x        = (const float*)d_in[0];
    const float* dw_w     = (const float*)d_in[1];
    const float* dw_bn_g  = (const float*)d_in[2];
    const float* dw_bn_b  = (const float*)d_in[3];
    const float* dw_bn_m  = (const float*)d_in[4];
    const float* dw_bn_v  = (const float*)d_in[5];
    const float* pw_w     = (const float*)d_in[6];
    const float* gn1_g    = (const float*)d_in[7];
    const float* gn1_b    = (const float*)d_in[8];
    const float* qkv_w    = (const float*)d_in[9];
    const float* qkv_b    = (const float*)d_in[10];
    const float* out_w    = (const float*)d_in[11];
    const float* out_b    = (const float*)d_in[12];
    const float* gn2_g    = (const float*)d_in[13];
    const float* gn2_b    = (const float*)d_in[14];
    const float* ffn1_w   = (const float*)d_in[15];
    const float* ffn1_b   = (const float*)d_in[16];
    const float* ffn2_w   = (const float*)d_in[17];
    const float* ffn2_b   = (const float*)d_in[18];
    const float* gnf_g    = (const float*)d_in[19];
    const float* gnf_b    = (const float*)d_in[20];
    const float* proj_w   = (const float*)d_in[21];
    const float* pj_bn_g  = (const float*)d_in[22];
    const float* pj_bn_b  = (const float*)d_in[23];
    const float* pj_bn_m  = (const float*)d_in[24];
    const float* pj_bn_v  = (const float*)d_in[25];

    // Workspace layout (total ~226 MB)
    char* base = (char*)d_ws;
    float* P      = (float*)(base);                              // 64 MB residual p
    float* G1     = (float*)(base + ((size_t)64  << 20));        // 64 MB k / ffn-hidden alias
    float* G2     = (float*)(base + ((size_t)128 << 20));        // 64 MB v / proj out
    bf16*  Xbf    = (bf16*) (base + ((size_t)192 << 20));        // 32 MB bf16 activations
    bf16*  Wbf    = (bf16*) (base + ((size_t)224 << 20));        // 512 KB bf16 weights
    float* scores = (float*)(base + ((size_t)224 << 20) + (512 << 10)); // 256 KB
    float* ctx    = scores + 65536;                              // 64 KB
    float* stats  = ctx + 16384;                                 // 128 B
    bf16*  X2bf   = (bf16*)G1;                                   // ffn hidden, aliases G1

    const dim3 blk(256);
    const dim3 gElem(65536);
    const dim3 gGemm256(SS / 128, 256 / 128);
    const dim3 gGemm512(SS / 128, 512 / 128);

    // 1. depthwise conv + BN + SiLU -> Xbf (patch column layout)
    dwconv_bn_silu_kernel<<<gElem, blk, 0, stream>>>(x, dw_w, dw_bn_g, dw_bn_b,
                                                     dw_bn_m, dw_bn_v, Xbf);
    // 2. pointwise conv: P = pw_w * Xbf
    f2bf_kernel<<<dim3((256 * 256 + 255) / 256), blk, 0, stream>>>(pw_w, Wbf, 256 * 256);
    gemm_bf16_wmma<0, 0><<<gGemm256, blk, 0, stream>>>(Wbf, Xbf, nullptr, P, 256, 256);

    for (int i = 0; i < 2; ++i) {
        // ---- linear attention ----
        zero_kernel<<<dim3(1), dim3(32), 0, stream>>>(stats, 32);
        gn_stats_kernel<<<dim3(16, 64), blk, 0, stream>>>(P, stats);
        gn_norm_kernel<<<gElem, blk, 0, stream>>>(P, stats, gn1_g + i * 256, gn1_b + i * 256, Xbf);

        f2bf_kernel<<<dim3((513 * 256 + 255) / 256), blk, 0, stream>>>(qkv_w + i * 513 * 256, Wbf, 513 * 256);
        // k = rows 1..256, v = rows 257..512 (relu fused)
        gemm_bf16_wmma<0, 0><<<gGemm256, blk, 0, stream>>>(Wbf + 256,       Xbf, qkv_b + i * 513 + 1,   G1, 256, 256);
        gemm_bf16_wmma<0, 2><<<gGemm256, blk, 0, stream>>>(Wbf + 257 * 256, Xbf, qkv_b + i * 513 + 257, G2, 256, 256);

        qsoftmax_kernel<<<dim3(64), blk, 0, stream>>>(Xbf, Wbf /*row 0*/, qkv_b + i * 513, scores);
        ctx_kernel<<<dim3(64, 32), blk, 0, stream>>>(G1, scores, ctx);
        vmul_kernel<<<gElem, blk, 0, stream>>>(G2, ctx, Xbf);

        f2bf_kernel<<<dim3((256 * 256 + 255) / 256), blk, 0, stream>>>(out_w + i * 256 * 256, Wbf, 256 * 256);
        // p += out_w * (relu(v)*ctx) + out_b
        gemm_bf16_wmma<1, 0><<<gGemm256, blk, 0, stream>>>(Wbf, Xbf, out_b + i * 256, P, 256, 256);

        // ---- FFN ----
        zero_kernel<<<dim3(1), dim3(32), 0, stream>>>(stats, 32);
        gn_stats_kernel<<<dim3(16, 64), blk, 0, stream>>>(P, stats);
        gn_norm_kernel<<<gElem, blk, 0, stream>>>(P, stats, gn2_g + i * 256, gn2_b + i * 256, Xbf);

        f2bf_kernel<<<dim3((512 * 256 + 255) / 256), blk, 0, stream>>>(ffn1_w + i * 512 * 256, Wbf, 512 * 256);
        // hidden = SiLU(ffn1 * x + b), bf16 out
        gemm_bf16_wmma<2, 1><<<gGemm512, blk, 0, stream>>>(Wbf, Xbf, ffn1_b + i * 512, X2bf, 512, 256);

        f2bf_kernel<<<dim3((256 * 512 + 255) / 256), blk, 0, stream>>>(ffn2_w + i * 256 * 512, Wbf, 256 * 512);
        // p += ffn2 * hidden + b
        gemm_bf16_wmma<1, 0><<<gGemm256, blk, 0, stream>>>(Wbf, X2bf, ffn2_b + i * 256, P, 256, 512);
    }

    // final GN -> proj conv -> fold + BN
    zero_kernel<<<dim3(1), dim3(32), 0, stream>>>(stats, 32);
    gn_stats_kernel<<<dim3(16, 64), blk, 0, stream>>>(P, stats);
    gn_norm_kernel<<<gElem, blk, 0, stream>>>(P, stats, gnf_g, gnf_b, Xbf);

    f2bf_kernel<<<dim3((256 * 256 + 255) / 256), blk, 0, stream>>>(proj_w, Wbf, 256 * 256);
    gemm_bf16_wmma<0, 0><<<gGemm256, blk, 0, stream>>>(Wbf, Xbf, nullptr, G2, 256, 256);

    bnfold_kernel<<<gElem, blk, 0, stream>>>(G2, pj_bn_g, pj_bn_b, pj_bn_m, pj_bn_v,
                                             (float*)d_out);
}
